// Similarity_Loss_group_deasease_4183298146522
// MI455X (gfx1250) — matrix operand order repaired and checked by reading, not compiled, not run on hardware
//
#include <hip/hip_runtime.h>
#include <hip/hip_bf16.h>
#include <hip/hip_fp16.h>

typedef __attribute__((ext_vector_type(16))) _Float16 v16h;
typedef __attribute__((ext_vector_type(8)))  _Float16 v8h;
typedef __attribute__((ext_vector_type(8)))  float    v8f;

#define BM 128
#define BN 128
#define NSPLIT 8

// ---------------------------------------------------------------------------
// Kernel 1: fp32 -> fp16 conversion + row squared-norms for both matrices.
// grid = (B, 2), block = 256. blockIdx.y selects text(0)/image(1).
// ---------------------------------------------------------------------------
__global__ __launch_bounds__(256) void prep_kernel(
    const float* __restrict__ text, const float* __restrict__ image,
    _Float16* __restrict__ tH, _Float16* __restrict__ mH,
    float* __restrict__ tn, float* __restrict__ mn, int Dd)
{
    const int row   = blockIdx.x;
    const int which = blockIdx.y;
    const float* src = which ? image : text;
    _Float16*    dst = which ? mH : tH;
    float*       nrm = which ? mn : tn;

    float ssq = 0.f;
    for (int c = threadIdx.x; c < Dd; c += 256) {
        float x = src[(size_t)row * Dd + c];
        dst[(size_t)row * Dd + c] = (_Float16)x;
        ssq += x * x;
    }
#pragma unroll
    for (int off = 16; off > 0; off >>= 1) ssq += __shfl_xor(ssq, off, 32);

    __shared__ float part[8];
    if ((threadIdx.x & 31) == 0) part[threadIdx.x >> 5] = ssq;
    __syncthreads();
    if (threadIdx.x == 0) {
        float t = 0.f;
#pragma unroll
        for (int w = 0; w < 8; ++w) t += part[w];
        nrm[row] = t;
    }
}

// ---------------------------------------------------------------------------
// Kernel 2: fused WMMA GEMM (T @ M^T) + distance/sim epilogue + per-row
// num/den accumulation. grid = (B/BM, NSPLIT), block = 256 (8 wave32 waves).
// A-stripe (128 x D fp16) staged once into LDS via async global->LDS DMA
// (CDNA5 GLOBAL_LOAD_ASYNC_TO_LDS_B128, ASYNCcnt); B frags streamed from
// global. Deterministic: no atomics; per-wave LDS slots summed in fixed order.
// ---------------------------------------------------------------------------
__global__ __launch_bounds__(256, 1) void gemm_loss_kernel(
    const _Float16* __restrict__ tH, const _Float16* __restrict__ mH,
    const float* __restrict__ tn, const float* __restrict__ mn,
    const int* __restrict__ groups,
    float* __restrict__ num, float* __restrict__ den,
    int Bn, int Dd, int lda)
{
    extern __shared__ char smem_raw[];
    _Float16* As    = (_Float16*)smem_raw;
    float*    sTn   = (float*)(smem_raw + (size_t)BM * lda * sizeof(_Float16));
    int*      sGrp  = (int*)(sTn + BM);
    float*    sNum  = (float*)(sGrp + BM);   // [4][BM] per wave-column
    float*    sDen  = sNum + 4 * BM;

    const int tid   = threadIdx.x;
    const int lane  = tid & 31;
    const int wave  = tid >> 5;
    const int wm    = wave >> 2;    // 0..1 : 64-row half of the stripe
    const int wn    = wave & 3;     // 0..3 : 32-col slice of the N tile
    const int half  = lane >> 4;    // hi/lo 16-lane group
    const int l16   = lane & 15;
    const int m0    = blockIdx.x * BM;
    const int split = blockIdx.y;

    // ---- stage A stripe (BM x Dd halves) into LDS via async DMA ----
    // Each lane issues a 16B global->LDS transfer; no VGPR round-trip.
    const int c8n = Dd >> 3;
    for (int i = tid; i < BM * c8n; i += 256) {
        int row = i / c8n;
        int c8  = i - row * c8n;
        unsigned loff =
            (unsigned)((const char*)(As + row * lda + c8 * 8) - smem_raw);
        unsigned long long gaddr =
            (unsigned long long)(tH + (size_t)(m0 + row) * Dd + c8 * 8);
        asm volatile("global_load_async_to_lds_b128 %0, %1, off"
                     :: "v"(loff), "v"(gaddr) : "memory");
    }
#if __has_builtin(__builtin_amdgcn_s_wait_asynccnt)
    __builtin_amdgcn_s_wait_asynccnt(0);
#else
    asm volatile("s_wait_asynccnt 0x0" ::: "memory");
#endif
    if (tid < BM) {
        sTn[tid]  = tn[m0 + tid];
        sGrp[tid] = groups[m0 + tid];
    }
    __syncthreads();

    float numAcc[4][8], denAcc[4][8];
#pragma unroll
    for (int mt = 0; mt < 4; ++mt)
#pragma unroll
        for (int v = 0; v < 8; ++v) { numAcc[mt][v] = 0.f; denAcc[mt][v] = 0.f; }

    const int tilesPerSplit = (Bn / BN) / NSPLIT;
    const int nBeg = split * tilesPerSplit * BN;
    const int nEnd = nBeg + tilesPerSplit * BN;

    for (int n0 = nBeg; n0 < nEnd; n0 += BN) {
        const v8f vzero = {};
        v8f acc[4][2];
#pragma unroll
        for (int mt = 0; mt < 4; ++mt)
#pragma unroll
            for (int nt = 0; nt < 2; ++nt) acc[mt][nt] = vzero;

        for (int k0 = 0; k0 < Dd; k0 += 32) {
            union F { v16h v; v8h h[2]; };
            F a[4], b[2];
            // A frag 16x32 f16 (ISA layout: lanes0-15 K0-7|K16-23, lanes16-31 K8-15|K24-31)
#pragma unroll
            for (int mt = 0; mt < 4; ++mt) {
                const _Float16* p = As + (wm * 64 + mt * 16 + l16) * lda + k0 + 8 * half;
                a[mt].h[0] = *(const v8h*)p;
                a[mt].h[1] = *(const v8h*)(p + 16);
            }
            // B frag 32x16 f16: lane = col, lanes0-15 K0-7|K16-23, lanes16-31 K8-15|K24-31
#pragma unroll
            for (int nt = 0; nt < 2; ++nt) {
                const _Float16* q = mH + (size_t)(n0 + wn * 32 + nt * 16 + l16) * Dd
                                       + k0 + 8 * half;
                b[nt].h[0] = *(const v8h*)q;
                b[nt].h[1] = *(const v8h*)(q + 16);
            }
#pragma unroll
            for (int mt = 0; mt < 4; ++mt)
#pragma unroll
                for (int nt = 0; nt < 2; ++nt)
                    acc[mt][nt] = __builtin_amdgcn_wmma_f32_16x16x32_f16(
                        false, a[mt].v, false, b[nt].v,
                        (short)0, acc[mt][nt], false, false);
        }

        // ---- fused epilogue: dist -> sim -> masked row accumulation ----
#pragma unroll
        for (int nt = 0; nt < 2; ++nt) {
            const int   ncol = n0 + wn * 32 + nt * 16 + l16;
            const float mnv  = mn[ncol];
            const int   gcol = groups[ncol];
#pragma unroll
            for (int mt = 0; mt < 4; ++mt) {
                const int rbase = wm * 64 + mt * 16 + 8 * half;
#pragma unroll
                for (int v = 0; v < 8; ++v) {
                    const int r   = rbase + v;
                    float     dot = acc[mt][nt][v];
                    float sq  = fmaxf(sTn[r] + mnv - 2.0f * dot, 0.0f);
                    float sim = __expf(-sqrtf(sq));
                    bool same = (sGrp[r] == gcol);
                    numAcc[mt][v] += same ? sim : 0.0f;
                    denAcc[mt][v] += same ? 0.0f : sim;
                }
            }
        }
    }

    // ---- reduce over the 16 lanes sharing each row (halves kept separate) ----
#pragma unroll
    for (int mt = 0; mt < 4; ++mt)
#pragma unroll
        for (int v = 0; v < 8; ++v) {
            float n = numAcc[mt][v], d = denAcc[mt][v];
#pragma unroll
            for (int off = 8; off > 0; off >>= 1) {
                n += __shfl_xor(n, off, 32);
                d += __shfl_xor(d, off, 32);
            }
            if (l16 == 0) {
                int r = wm * 64 + mt * 16 + 8 * half + v;
                sNum[wn * BM + r] = n;
                sDen[wn * BM + r] = d;
            }
        }
    __syncthreads();
    if (tid < BM) {
        float n = 0.f, d = 0.f;
#pragma unroll
        for (int w = 0; w < 4; ++w) { n += sNum[w * BM + tid]; d += sDen[w * BM + tid]; }
        num[(size_t)split * Bn + m0 + tid] = n;
        den[(size_t)split * Bn + m0 + tid] = d;
    }
}

// ---------------------------------------------------------------------------
// Kernel 3: combine NSPLIT partials per row, per-row loss, mean. grid = 1.
// ---------------------------------------------------------------------------
__global__ __launch_bounds__(256) void loss_kernel(
    const float* __restrict__ num, const float* __restrict__ den,
    float* __restrict__ out, int Bn)
{
    __shared__ float red[256];
    float s = 0.f;
    for (int i = threadIdx.x; i < Bn; i += 256) {
        float n = 0.f, d = 0.f;
#pragma unroll
        for (int sp = 0; sp < NSPLIT; ++sp) {
            n += num[(size_t)sp * Bn + i];
            d += den[(size_t)sp * Bn + i];
        }
        if (n > 0.f && d > 0.f) s -= __logf(n / d);
    }
    red[threadIdx.x] = s;
    __syncthreads();
    for (int off = 128; off > 0; off >>= 1) {
        if (threadIdx.x < off) red[threadIdx.x] += red[threadIdx.x + off];
        __syncthreads();
    }
    if (threadIdx.x == 0) out[0] = red[0] / (float)Bn;
}

// ---------------------------------------------------------------------------
extern "C" void kernel_launch(void* const* d_in, const int* in_sizes, int n_in,
                              void* d_out, int out_size, void* d_ws, size_t ws_size,
                              hipStream_t stream) {
    const float* text   = (const float*)d_in[0];
    const float* image  = (const float*)d_in[1];
    const int*   groups = (const int*)d_in[2];
    float*       out    = (float*)d_out;

    const int Bn = in_sizes[2];           // 8192
    const int Dd = in_sizes[0] / Bn;      // 512

    // workspace layout
    char*  ws       = (char*)d_ws;
    size_t embBytes = (size_t)Bn * Dd * sizeof(_Float16);
    _Float16* tH = (_Float16*)ws;
    _Float16* mH = (_Float16*)(ws + embBytes);
    float* tn  = (float*)(ws + 2 * embBytes);
    float* mn  = tn + Bn;
    float* num = mn + Bn;
    float* den = num + (size_t)NSPLIT * Bn;

    prep_kernel<<<dim3(Bn, 2), 256, 0, stream>>>(text, image, tH, mH, tn, mn, Dd);

    const int lda = Dd + 8;  // pad 8 halves -> +4 dword banks per row, conflict-free
    size_t smem = (size_t)BM * lda * sizeof(_Float16)   // As
                + BM * sizeof(float)                    // sTn
                + BM * sizeof(int)                      // sGrp
                + 2 * 4 * BM * sizeof(float);           // sNum + sDen
    hipFuncSetAttribute((const void*)gemm_loss_kernel,
                        hipFuncAttributeMaxDynamicSharedMemorySize, (int)smem);
    gemm_loss_kernel<<<dim3(Bn / BM, NSPLIT), 256, smem, stream>>>(
        tH, mH, tn, mn, groups, num, den, Bn, Dd, lda);

    loss_kernel<<<1, 256, 0, stream>>>(num, den, out, Bn);
}
